// FusedScatteringAttention_87625922773412
// MI455X (gfx1250) — compile-verified, matrix operand order checked by reading, not run on hardware
//
#include <hip/hip_runtime.h>
#include <hip/hip_bf16.h>
#include <math.h>

// ---------------------------------------------------------------------------
// FusedScatteringAttention for MI455X (gfx1250, wave32, WMMA 16x16x32 f16).
// B=2, L=2048, D=1024, H=16, hd=64.
// Pipeline:
//   0) f32->f16 convert of x, qkv_w, out_w (one cheap streaming pass)
//   1) gate softmax
//   2) QKV GEMM      (async-to-LDS staging + WMMA)
//   3) flash attention (WMMA QK^T + online softmax + WMMA PV, async staging)
//   4) output GEMM   (+bias, fp32 out)
// ---------------------------------------------------------------------------

typedef _Float16 v8h  __attribute__((ext_vector_type(8)));
typedef _Float16 v16h __attribute__((ext_vector_type(16)));
typedef float    v4f  __attribute__((ext_vector_type(4)));
typedef float    v8f  __attribute__((ext_vector_type(8)));

#define BATCH 2
#define SEQ   2048
#define DMODEL 1024
#define NHEAD 16
#define HDIM  64

__device__ inline v8f wmma_f16(v16h a, v16h b, v8f c) {
  return __builtin_amdgcn_wmma_f32_16x16x32_f16(
      false, a, false, b, (short)0, c, false, false);
}

// CDNA5 async copy: 16B global -> LDS, tracked by ASYNCcnt (no VGPR data).
// Generic LDS pointers carry the wave-relative LDS address in addr[31:0]
// (ISA 10.2 aperture table), so truncation yields the ds address.
__device__ inline void async_copy16(void* lds_dst, const void* gsrc) {
  const unsigned lds_off = (unsigned)(uintptr_t)lds_dst;
  const unsigned long long ga = (unsigned long long)(uintptr_t)gsrc;
  asm volatile("global_load_async_to_lds_b128 %0, %1, off"
               :
               : "v"(lds_off), "v"(ga)
               : "memory");
}
__device__ inline void wait_async0() {
  asm volatile("s_wait_asynccnt 0x0" ::: "memory");
}

// Load a 16x32 f16 A-fragment (or column-major B-fragment) from an LDS tile.
// lane l (m = l&15, h = l>>4) holds k = {8h..8h+7, 16+8h..16+8h+7}:
// two contiguous 8-half chunks -> two ds_read_b128 per fragment.
__device__ inline v16h load_frag16(const _Float16* __restrict__ tile, int lda,
                                   int kbase, int lane) {
  const int m = lane & 15;
  const int h = lane >> 4;
  const _Float16* p = tile + m * lda + kbase + 8 * h;
  v8h lo = *(const v8h*)(p);
  v8h hi = *(const v8h*)(p + 16);
  v16h r;
#pragma unroll
  for (int i = 0; i < 8; ++i) { r[i] = lo[i]; r[i + 8] = hi[i]; }
  return r;
}

// ---------------------------------------------------------------------------
// f32 -> f16 streaming convert (n multiple of 2048).
// ---------------------------------------------------------------------------
__global__ __launch_bounds__(256) void cvt_kernel(const float* __restrict__ in,
                                                  _Float16* __restrict__ outh,
                                                  int n) {
  const int i = (blockIdx.x * 256 + threadIdx.x) * 8;
  if (i < n) {
    v4f a = *(const v4f*)(in + i);
    v4f b = *(const v4f*)(in + i + 4);
    v8h hh;
#pragma unroll
    for (int e = 0; e < 4; ++e) {
      hh[e] = (_Float16)a[e];
      hh[e + 4] = (_Float16)b[e];
    }
    *(v8h*)(outh + i) = hh;
  }
}

// ---------------------------------------------------------------------------
// Gate: softmax over L of sqrt(re^2+im^2+eps)/temp, times gate_scale.
// ---------------------------------------------------------------------------
__global__ __launch_bounds__(256) void gate_kernel(
    const float* __restrict__ G, const float* __restrict__ gscale,
    const float* __restrict__ ltemp, float* __restrict__ gate) {
  __shared__ float red[256];
  const int b = blockIdx.x, tid = threadIdx.x;
  const float invt = __expf(-ltemp[0]);
  float z[8];
  float mx = -3.0e38f;
#pragma unroll
  for (int i = 0; i < 8; ++i) {
    const int l = i * 256 + tid;
    const float re = G[((size_t)b * SEQ + l) * 2 + 0];
    const float im = G[((size_t)b * SEQ + l) * 2 + 1];
    z[i] = sqrtf(re * re + im * im + 1e-7f) * invt;
    mx = fmaxf(mx, z[i]);
  }
  red[tid] = mx;
  __syncthreads();
  for (int s = 128; s > 0; s >>= 1) {
    if (tid < s) red[tid] = fmaxf(red[tid], red[tid + s]);
    __syncthreads();
  }
  const float zmax = red[0];
  __syncthreads();
  float sum = 0.f;
#pragma unroll
  for (int i = 0; i < 8; ++i) sum += __expf(z[i] - zmax);
  red[tid] = sum;
  __syncthreads();
  for (int s = 128; s > 0; s >>= 1) {
    if (tid < s) red[tid] += red[tid + s];
    __syncthreads();
  }
  const float sc = gscale[0] / red[0];
#pragma unroll
  for (int i = 0; i < 8; ++i)
    gate[(size_t)b * SEQ + i * 256 + tid] = __expf(z[i] - zmax) * sc;
}

// ---------------------------------------------------------------------------
// GEMM: C(M,N) = A(M,K) * W(N,K)^T, A and W both f16.
// W row-major == WMMA B column-major (free transpose).
// 128x128 block tile, k-step 32, 8 waves (4x2), wave tile 32x64.
// Double-buffered LDS filled by global_load_async_to_lds_b128; one barrier
// per k-step.  EPI 0: scatter f16 q/k/v (B,H,L,hd).  EPI 1: fp32+bias out.
// ---------------------------------------------------------------------------
#define TM 128
#define TN 128
#define TK 32
#define LDT 40  // 32+8 halfs pad: 80B row stride, keeps 16B alignment

template <int EPI>
__global__ __launch_bounds__(256) void gemm_kernel(
    const _Float16* __restrict__ Ah, const _Float16* __restrict__ Bh,
    int M, int N, int K,
    _Float16* __restrict__ qq, _Float16* __restrict__ qk,
    _Float16* __restrict__ qv,
    const float* __restrict__ bias, float* __restrict__ outf) {
  __shared__ _Float16 As[2][TM * LDT];
  __shared__ _Float16 Bs[2][TN * LDT];

  const int tid = threadIdx.x;
  const int lane = tid & 31;
  const int wave = tid >> 5;
  const int wm = wave >> 1;  // 0..3
  const int wn = wave & 1;   // 0..1
  const int m0 = blockIdx.y * TM;
  const int n0 = blockIdx.x * TN;

  const int lr = tid >> 1;        // 0..127: tile row
  const int lc = (tid & 1) * 16;  // 0/16: k-chunk (16 halfs = 2 x b128)

  auto async_stage = [&](int buf, int kt) {
    const int kb = kt * TK + lc;
    const _Float16* ga = Ah + (size_t)(m0 + lr) * K + kb;
    const _Float16* gb = Bh + (size_t)(n0 + lr) * K + kb;
    _Float16* la = &As[buf][lr * LDT + lc];
    _Float16* lb = &Bs[buf][lr * LDT + lc];
    async_copy16(la, ga);
    async_copy16(la + 8, ga + 8);
    async_copy16(lb, gb);
    async_copy16(lb + 8, gb + 8);
  };

  v8f acc[2][4];
#pragma unroll
  for (int mi = 0; mi < 2; ++mi)
#pragma unroll
    for (int ni = 0; ni < 4; ++ni)
#pragma unroll
      for (int e = 0; e < 8; ++e) acc[mi][ni][e] = 0.f;

  const int KT = K / TK;
  async_stage(0, 0);
  wait_async0();
  __syncthreads();

  for (int kt = 0; kt < KT; ++kt) {
    const int cur = kt & 1;
    if (kt + 1 < KT) async_stage(1 - cur, kt + 1);  // fill other buffer

    v16h af0 = load_frag16(&As[cur][(wm * 32 + 0) * LDT], LDT, 0, lane);
    v16h af1 = load_frag16(&As[cur][(wm * 32 + 16) * LDT], LDT, 0, lane);
#pragma unroll
    for (int ni = 0; ni < 4; ++ni) {
      v16h bf = load_frag16(&Bs[cur][(wn * 64 + ni * 16) * LDT], LDT, 0, lane);
      acc[0][ni] = wmma_f16(af0, bf, acc[0][ni]);
      acc[1][ni] = wmma_f16(af1, bf, acc[1][ni]);
    }
    if (kt + 1 < KT) wait_async0();  // my async fills landed
    __syncthreads();                 // everyone done reading cur + fills visible
  }

  const int hf = lane >> 4;
  const int cl = lane & 15;
#pragma unroll
  for (int mi = 0; mi < 2; ++mi) {
#pragma unroll
    for (int ni = 0; ni < 4; ++ni) {
#pragma unroll
      for (int r = 0; r < 8; ++r) {
        const int m = m0 + wm * 32 + mi * 16 + r + 8 * hf;
        const int n = n0 + wn * 64 + ni * 16 + cl;
        const float val = acc[mi][ni][r];
        if (EPI == 0) {
          const int which = n >> 10;
          const int rem = n & 1023;
          const int head = rem >> 6;
          const int dim = rem & 63;
          const int bb = m >> 11;
          const int ll = m & 2047;
          _Float16* dst = (which == 0) ? qq : ((which == 1) ? qk : qv);
          dst[(((size_t)(bb * NHEAD + head) * SEQ + ll) * HDIM) + dim] =
              (_Float16)val;
        } else {
          outf[(size_t)m * N + n] = val + bias[n];
        }
      }
    }
  }
}

// ---------------------------------------------------------------------------
// Flash attention: grid (L/128, B*H), 256 threads = 8 waves.
// Wave w owns query rows [q0+16w, q0+16w+16).  Key tiles of 64.
// Q/K tiles staged via async-to-LDS; V transposed through registers.
// ---------------------------------------------------------------------------
__global__ __launch_bounds__(256) void attn_kernel(
    const _Float16* __restrict__ q, const _Float16* __restrict__ k,
    const _Float16* __restrict__ v, const float* __restrict__ gate,
    _Float16* __restrict__ aout) {
  __shared__ _Float16 Qs[128 * 72];
  __shared__ _Float16 Ks[64 * 72];
  __shared__ _Float16 Vt[64 * 72];     // transposed: [dim][key]
  __shared__ _Float16 Ps[8][16 * 72];  // per-wave P staging (C -> A layout)

  const int tid = threadIdx.x;
  const int lane = tid & 31;
  const int wave = tid >> 5;
  const int bh = blockIdx.y;  // b*16 + h
  const int b = bh >> 4;
  const int h = bh & 15;
  const int q0 = blockIdx.x * 128;
  const size_t base = (size_t)bh * SEQ * HDIM;

  // ---- stage Q tile via async copies, build register-resident A-fragments
  {
    const int r = tid >> 1;
    const int c0 = (tid & 1) * 32;
    const _Float16* src = q + base + (size_t)(q0 + r) * HDIM + c0;
#pragma unroll
    for (int j = 0; j < 4; ++j)
      async_copy16(&Qs[r * 72 + c0 + 8 * j], src + 8 * j);
  }
  wait_async0();
  __syncthreads();
  v16h qf0 = load_frag16(&Qs[(wave * 16) * 72], 72, 0, lane);
  v16h qf1 = load_frag16(&Qs[(wave * 16) * 72], 72, 32, lane);

  v8f oacc[4];
  float rm[8], rl[8];
#pragma unroll
  for (int nt = 0; nt < 4; ++nt)
#pragma unroll
    for (int e = 0; e < 8; ++e) oacc[nt][e] = 0.f;
#pragma unroll
  for (int r = 0; r < 8; ++r) { rm[r] = -3.0e38f; rl[r] = 0.f; }

  const float scale = 0.125f;  // 1/sqrt(64)
  const int hf = lane >> 4;
  const int cl = lane & 15;

  for (int ktile = 0; ktile < SEQ / 64; ++ktile) {
    __syncthreads();  // previous iteration's K/V readers done
    {
      const int r = tid >> 2;
      const int c0 = (tid & 3) * 16;
      // K: async straight into LDS, row-major (key x hd)
      const _Float16* ksrc = k + base + (size_t)(ktile * 64 + r) * HDIM + c0;
      async_copy16(&Ks[r * 72 + c0], ksrc);
      async_copy16(&Ks[r * 72 + c0 + 8], ksrc + 8);
      // V: through registers with transpose (hd x key)
      const _Float16* vsrc = v + base + (size_t)(ktile * 64 + r) * HDIM + c0;
      v8h u0 = *(const v8h*)(vsrc);
      v8h u1 = *(const v8h*)(vsrc + 8);
#pragma unroll
      for (int j = 0; j < 8; ++j) {
        Vt[(c0 + j) * 72 + r] = u0[j];
        Vt[(c0 + 8 + j) * 72 + r] = u1[j];
      }
    }
    wait_async0();
    __syncthreads();

    // ---- S = Q * K^T  (K rows serve as column-major B)
    v8f sacc[4];
#pragma unroll
    for (int nt = 0; nt < 4; ++nt) {
#pragma unroll
      for (int e = 0; e < 8; ++e) sacc[nt][e] = 0.f;
      v16h bf0 = load_frag16(&Ks[(nt * 16) * 72], 72, 0, lane);
      sacc[nt] = wmma_f16(qf0, bf0, sacc[nt]);
      v16h bf1 = load_frag16(&Ks[(nt * 16) * 72], 72, 32, lane);
      sacc[nt] = wmma_f16(qf1, bf1, sacc[nt]);
    }

    // ---- online softmax (row r in VGPR index, cols across 16 lanes)
#pragma unroll
    for (int r = 0; r < 8; ++r) {
      const float s0 = sacc[0][r] * scale;
      const float s1 = sacc[1][r] * scale;
      const float s2 = sacc[2][r] * scale;
      const float s3 = sacc[3][r] * scale;
      float t = fmaxf(fmaxf(s0, s1), fmaxf(s2, s3));
#pragma unroll
      for (int msk = 1; msk <= 8; msk <<= 1)
        t = fmaxf(t, __shfl_xor(t, msk, 32));
      const float mnew = fmaxf(rm[r], t);
      const float alpha = __expf(rm[r] - mnew);
      rm[r] = mnew;
      const float p0 = __expf(s0 - mnew);
      const float p1 = __expf(s1 - mnew);
      const float p2 = __expf(s2 - mnew);
      const float p3 = __expf(s3 - mnew);
      _Float16* pp = &Ps[wave][(r + 8 * hf) * 72];
      pp[0 + cl] = (_Float16)p0;
      pp[16 + cl] = (_Float16)p1;
      pp[32 + cl] = (_Float16)p2;
      pp[48 + cl] = (_Float16)p3;
      float ps = p0 + p1 + p2 + p3;
#pragma unroll
      for (int msk = 1; msk <= 8; msk <<= 1) ps += __shfl_xor(ps, msk, 32);
      rl[r] = rl[r] * alpha + ps;
#pragma unroll
      for (int nt = 0; nt < 4; ++nt) oacc[nt][r] *= alpha;
    }

    // ---- O += P * V  (Vt rows serve as column-major B over dims)
#pragma unroll
    for (int ks = 0; ks < 2; ++ks) {
      v16h pf = load_frag16(Ps[wave], 72, 32 * ks, lane);
#pragma unroll
      for (int nt = 0; nt < 4; ++nt) {
        v16h vf = load_frag16(&Vt[(nt * 16) * 72], 72, 32 * ks, lane);
        oacc[nt] = wmma_f16(pf, vf, oacc[nt]);
      }
    }
  }

  // ---- epilogue: normalize, gate, store f16 (B,L,D)
#pragma unroll
  for (int r = 0; r < 8; ++r) {
    const int row = q0 + wave * 16 + r + 8 * hf;
    const float g = gate[(size_t)b * SEQ + row] / rl[r];
#pragma unroll
    for (int nt = 0; nt < 4; ++nt) {
      aout[((size_t)(b * SEQ + row)) * DMODEL + h * HDIM + nt * 16 + cl] =
          (_Float16)(oacc[nt][r] * g);
    }
  }
}

// ---------------------------------------------------------------------------
extern "C" void kernel_launch(void* const* d_in, const int* in_sizes, int n_in,
                              void* d_out, int out_size, void* d_ws,
                              size_t ws_size, hipStream_t stream) {
  const float* x = (const float*)d_in[0];
  const float* G = (const float*)d_in[1];
  const float* qkv_w = (const float*)d_in[2];
  const float* out_w = (const float*)d_in[3];
  const float* out_b = (const float*)d_in[4];
  const float* gate_scale = (const float*)d_in[5];
  const float* log_temp = (const float*)d_in[6];
  float* out = (float*)d_out;

  const size_t BLD = (size_t)BATCH * SEQ * DMODEL;  // 4M elems
  _Float16* qb = (_Float16*)d_ws;
  _Float16* kb = qb + BLD;
  _Float16* vb = kb + BLD;
  _Float16* ab = vb + BLD;            // attention out f16 (B,L,D)
  _Float16* xh = ab + BLD;            // x in f16
  _Float16* wqkvh = xh + BLD;         // qkv_w f16 (3D x D)
  _Float16* wouth = wqkvh + 3 * DMODEL * DMODEL;  // out_w f16 (D x D)
  float* gate = (float*)(wouth + DMODEL * DMODEL);

  // 0) f32 -> f16 converts (streaming, ~50MB total)
  cvt_kernel<<<dim3((int)(BLD / 2048)), dim3(256), 0, stream>>>(x, xh,
                                                                (int)BLD);
  cvt_kernel<<<dim3(3 * DMODEL * DMODEL / 2048), dim3(256), 0, stream>>>(
      qkv_w, wqkvh, 3 * DMODEL * DMODEL);
  cvt_kernel<<<dim3(DMODEL * DMODEL / 2048), dim3(256), 0, stream>>>(
      out_w, wouth, DMODEL * DMODEL);

  // 1) gate
  gate_kernel<<<dim3(BATCH), dim3(256), 0, stream>>>(G, gate_scale, log_temp,
                                                     gate);

  // 2) QKV: M = 4096, N = 3072, K = 1024
  gemm_kernel<0><<<dim3(3072 / TN, 4096 / TM), dim3(256), 0, stream>>>(
      xh, wqkvh, 4096, 3072, 1024, qb, kb, vb, nullptr, nullptr);

  // 3) flash attention
  attn_kernel<<<dim3(SEQ / 128, BATCH * NHEAD), dim3(256), 0, stream>>>(
      qb, kb, vb, gate, ab);

  // 4) output projection: M = 4096, N = 1024, K = 1024
  gemm_kernel<1><<<dim3(1024 / TN, 4096 / TM), dim3(256), 0, stream>>>(
      ab, wouth, 4096, 1024, 1024, nullptr, nullptr, nullptr, out_b, out);
}
